// MatchMambaBlock_Last_63720134803673
// MI455X (gfx1250) — compile-verified, matrix operand order checked by reading, not run on hardware
//
#include <hip/hip_runtime.h>
#include <math.h>

// ---------------------------------------------------------------------------
// Shapes (fixed by the reference)
// ---------------------------------------------------------------------------
#define L_      4096          // H*W*Dd
#define BL_     8192          // B * L
#define DIM_    256
#define DI_     256
#define NST_    8             // SSM state dim
#define DTR_    16
#define OUTC_   1200

// ---------------------------------------------------------------------------
// Workspace layout (bytes) — ~72 MB total
// ---------------------------------------------------------------------------
#define OFF_SEQ0H  ((size_t)0)            // BL*256 f16  (4 MB)
#define OFF_SEQ1H  ((size_t)4  << 20)     // BL*256 f16  (4 MB)
#define OFF_WTS    ((size_t)8  << 20)     // 536576 f16  (~1.05 MB, 2 MB reserved)
#define OFF_X      ((size_t)10 << 20)     // BL*256 f32  (8 MB)
#define OFF_Z      ((size_t)18 << 20)     // BL*256 f32  (8 MB)
#define OFF_XS     ((size_t)26 << 20)     // BL*256 f32  (8 MB)  per-direction, reused
#define OFF_XSH    ((size_t)34 << 20)     // BL*256 f16  (4 MB)
#define OFF_DBL    ((size_t)38 << 20)     // BL*32  f32  (1 MB)
#define OFF_DTH    ((size_t)39 << 20)     // BL*32  f16  (0.5 MB)
#define OFF_DT     ((size_t)40 << 20)     // BL*256 f32  (8 MB)
#define OFF_YF     ((size_t)48 << 20)     // BL*256 f32  (8 MB)
#define OFF_YB     ((size_t)56 << 20)     // BL*256 f32  (8 MB)
#define OFF_YH     ((size_t)64 << 20)     // BL*256 f16  (4 MB)  per-head, reused
#define OFF_OH     ((size_t)68 << 20)     // BL*256 f16  (4 MB)  per-head, reused

// f16-weight region (offsets in halves)
#define WH_INPROJ   0
#define WH_INPROJZ  65536
#define WH_XPROJ    131072
#define WH_XPROJB   139264
#define WH_DTPROJ   147456      // (256 x 32), K padded 16->32
#define WH_DTPROJB  155648
#define WH_OUTPROJ  163840
#define WH_CONV3D   229376      // (1200 x 256)
#define WH_TOTAL    536576

typedef __attribute__((ext_vector_type(16))) _Float16     v16h;
typedef __attribute__((ext_vector_type(8)))  _Float16     v8h;
typedef __attribute__((ext_vector_type(8)))  float        v8f;
typedef __attribute__((ext_vector_type(4)))  unsigned int v4u;
typedef __attribute__((ext_vector_type(8)))  int          v8i;
typedef __attribute__((ext_vector_type(4)))  int          v4i;

union FragU { v16h v; v8h h[2]; };

// ---------------------------------------------------------------------------
// WMMA 16x32 f16 fragment load (ISA 7.12.2 layout):
// lanes 0-15 hold row M=lane, halves 0..7 = K[kh*8 .. kh*8+7],
// halves 8..15 = K[16+kh*8 .. 16+kh*8+7], where kh = lane>>4.
// Row-major source, contiguous K => two 16-byte loads.
// ---------------------------------------------------------------------------
__device__ __forceinline__ v16h load_frag(const _Float16* __restrict__ base,
                                          int row0, int ld, int k0, int lane) {
  const int r  = row0 + (lane & 15);
  const int kh = (lane >> 4) & 1;
  const _Float16* p = base + (size_t)r * ld + k0 + kh * 8;
  FragU f;
  f.h[0] = *(const v8h*)(p);
  f.h[1] = *(const v8h*)(p + 16);
  return f.v;
}

// ---------------------------------------------------------------------------
// LDS-staged WMMA GEMM for the fat projections (K = 256).
// Block = 128 threads (4 waves). Each block owns a (128 x NT*16) C tile:
//   - wave w computes rows m0 = (blockIdx.x*4+w)*32 .. +31 (two A fragments)
//   - all waves share one B tile (NT*16 rows x K) staged in LDS by the
//     Tensor Data Mover (one TENSOR_LOAD_TO_LDS, s_wait_tensorcnt, barrier)
// EPI: 0 = store f32
//      2 = acc + skip(input0) -> f16 (out_proj)
//      3 = acc + aux[n] scattered to (B, OUTC, L) layout (conv3d)
// ---------------------------------------------------------------------------
template<int NT, int EPI>
__global__ __launch_bounds__(128)
void wmma_gemm_lds_k(const _Float16* __restrict__ A, int lda,
                     const _Float16* __restrict__ Bw, int Kdim, int Ntot,
                     void* __restrict__ outp, int ldo,
                     const float* __restrict__ aux) {
  __shared__ _Float16 ldsB[NT * 16 * 256];
  const int lane = threadIdx.x & 31;
  const int wave = threadIdx.x >> 5;
  const int n0   = blockIdx.y * (NT * 16);

#if __has_builtin(__builtin_amdgcn_tensor_load_to_lds)
  if (threadIdx.x < 32) {
    // Build a 2-D Tensor DMA descriptor (D#) per CDNA5 ISA:
    // load tile (rows = NT*16, cols = Kdim halves) from row-major weights.
    const unsigned long long ga =
        (unsigned long long)(uintptr_t)(Bw + (size_t)n0 * Kdim);
    const unsigned ldsOff = (unsigned)(uintptr_t)&ldsB[0];
    v4u g0;
    g0[0] = 1u;                                  // count=1, user descriptor
    g0[1] = ldsOff;                              // lds_addr (bytes)
    g0[2] = (unsigned)(ga & 0xFFFFFFFFu);        // global_addr[31:0]
    g0[3] = (unsigned)((ga >> 32) & 0x01FFFFFFu) // global_addr[56:32]
            | 0x80000000u;                       // type=2 ("image")
    v8i g1;
    g1[0] = 0x00010000;                                    // data_size = 2 bytes
    g1[1] = (Kdim & 0xFFFF) << 16;                         // tensor_dim0[15:0]
    g1[2] = ((Kdim >> 16) & 0xFFFF) |
            ((Ntot & 0xFFFF) << 16);                       // dim0 hi | tensor_dim1 lo
    g1[3] = ((Ntot >> 16) & 0xFFFF) |
            ((Kdim & 0xFFFF) << 16);                       // dim1 hi | tile_dim0
    g1[4] = NT * 16;                                       // tile_dim1 (tile_dim2=0)
    g1[5] = Kdim;                                          // tensor_dim0_stride[31:0]
    g1[6] = 0;
    g1[7] = 0;
    v4i z4 = {0, 0, 0, 0};
#if __clang_major__ >= 23
    v8i z8 = {0, 0, 0, 0, 0, 0, 0, 0};
    __builtin_amdgcn_tensor_load_to_lds(g0, g1, z4, z4, z8, 0);
#else
    __builtin_amdgcn_tensor_load_to_lds(g0, g1, z4, z4, 0);
#endif
    __builtin_amdgcn_s_wait_tensorcnt(0);
  }
#else
  for (int i = threadIdx.x; i < NT * 16 * Kdim / 8; i += 128)
    ((v8h*)ldsB)[i] = ((const v8h*)(Bw + (size_t)n0 * Kdim))[i];
#endif
  __syncthreads();

  const int m0 = (blockIdx.x * 4 + wave) * 32;
  v8f acc[2][NT] = {};
  for (int k0 = 0; k0 < Kdim; k0 += 32) {
    v16h a0 = load_frag(A, m0,      lda, k0, lane);
    v16h a1 = load_frag(A, m0 + 16, lda, k0, lane);
    if (k0 + 32 < Kdim) {
      __builtin_prefetch(A + (size_t)(m0 + (lane & 15)) * lda + k0 + 32, 0, 3);
    }
#pragma unroll
    for (int t = 0; t < NT; ++t) {
      v16h b = load_frag(ldsB, t * 16, Kdim, k0, lane);   // ds loads
      acc[0][t] = __builtin_amdgcn_wmma_f32_16x16x32_f16(
          false, a0, false, b, (short)0, acc[0][t], false, false);
      acc[1][t] = __builtin_amdgcn_wmma_f32_16x16x32_f16(
          false, a1, false, b, (short)0, acc[1][t], false, false);
    }
  }

  // C/D layout: VGPR r, lanes 0-15 -> M=r, lanes 16-31 -> M=8+r; N = lane&15.
  const int ncol = lane & 15;
#pragma unroll
  for (int s = 0; s < 2; ++s) {
    const int mbase = m0 + s * 16 + ((lane >> 4) << 3);
#pragma unroll
    for (int t = 0; t < NT; ++t) {
      const int n = n0 + t * 16 + ncol;
#pragma unroll
      for (int r = 0; r < 8; ++r) {
        const int row = mbase + r;
        const float v = acc[s][t][r];
        if (EPI == 0) {
          ((float*)outp)[(size_t)row * ldo + n] = v;
        } else if (EPI == 2) {
          const int b = row >> 12, l = row & (L_ - 1);
          const float sk = aux[((size_t)b * DIM_ + n) * L_ + l];
          ((_Float16*)outp)[(size_t)row * ldo + n] = (_Float16)(v + sk);
        } else {
          const int b = row >> 12, l = row & (L_ - 1);
          ((float*)outp)[((size_t)b * OUTC_ + n) * L_ + l] = v + aux[n];
        }
      }
    }
  }
}

// ---------------------------------------------------------------------------
// Direct-global per-wave WMMA GEMM for the skinny projections.
// EPI: 0 = store f32; 1 = softplus(acc+aux[n]) (dtproj)
// ---------------------------------------------------------------------------
template<int NT, int EPI>
__global__ __launch_bounds__(32)
void wmma_gemm_k(const _Float16* __restrict__ A, int lda,
                 const _Float16* __restrict__ Bw, int ldb, int Kdim,
                 void* __restrict__ outp, int ldo,
                 const float* __restrict__ aux) {
  const int lane = threadIdx.x;
  const int m0   = blockIdx.x * 16;
  const int ng0  = blockIdx.y * NT;

  v8f acc[NT] = {};
  for (int k0 = 0; k0 < Kdim; k0 += 32) {
    v16h a = load_frag(A, m0, lda, k0, lane);
#pragma unroll
    for (int t = 0; t < NT; ++t) {
      v16h b = load_frag(Bw, (ng0 + t) * 16, ldb, k0, lane);
      acc[t] = __builtin_amdgcn_wmma_f32_16x16x32_f16(
          false, a, false, b, (short)0, acc[t], false, false);
    }
  }
  const int mbase = m0 + ((lane >> 4) << 3);
  const int ncol  = lane & 15;
#pragma unroll
  for (int t = 0; t < NT; ++t) {
    const int n = (ng0 + t) * 16 + ncol;
#pragma unroll
    for (int r = 0; r < 8; ++r) {
      const int row = mbase + r;
      const float v = acc[t][r];
      if (EPI == 0) {
        ((float*)outp)[(size_t)row * ldo + n] = v;
      } else {
        const float x = v + aux[n];
        ((float*)outp)[(size_t)row * ldo + n] = (x > 20.f) ? x : log1pf(__expf(x));
      }
    }
  }
}

// ---------------------------------------------------------------------------
// LayerNorm over DIM with (b,c,l) -> (b*L+l, c) transpose; f16 output.
// ---------------------------------------------------------------------------
__global__ __launch_bounds__(32)
void ln_rows_k(const float* __restrict__ in, const float* __restrict__ w,
               const float* __restrict__ bia, _Float16* __restrict__ outh) {
  const int row = blockIdx.x;            // b*L + l
  const int b = row >> 12, l = row & (L_ - 1);
  const int lane = threadIdx.x;
  float v[8], s = 0.f;
#pragma unroll
  for (int j = 0; j < 8; ++j) {
    const int c = lane + j * 32;
    v[j] = in[((size_t)b * DIM_ + c) * L_ + l];
    s += v[j];
  }
#pragma unroll
  for (int m = 16; m >= 1; m >>= 1) s += __shfl_xor(s, m, 32);
  const float mean = s * (1.f / 256.f);
  float vs = 0.f;
#pragma unroll
  for (int j = 0; j < 8; ++j) { const float d = v[j] - mean; vs += d * d; }
#pragma unroll
  for (int m = 16; m >= 1; m >>= 1) vs += __shfl_xor(vs, m, 32);
  const float inv = rsqrtf(vs * (1.f / 256.f) + 1e-5f);
#pragma unroll
  for (int j = 0; j < 8; ++j) {
    const int c = lane + j * 32;
    outh[(size_t)row * DIM_ + c] = (_Float16)((v[j] - mean) * inv * w[c] + bia[c]);
  }
}

// ---------------------------------------------------------------------------
// f32 -> f16 weight repack (dtproj K padded 16 -> 32 with zeros)
// ---------------------------------------------------------------------------
__global__ void cvt_w_k(const float* __restrict__ w6,  const float* __restrict__ w7,
                        const float* __restrict__ w12, const float* __restrict__ w13,
                        const float* __restrict__ w14, const float* __restrict__ w16,
                        const float* __restrict__ w23, const float* __restrict__ w24,
                        _Float16* __restrict__ wh) {
  const int i = blockIdx.x * blockDim.x + threadIdx.x;
  if (i >= WH_TOTAL) return;
  float v;
  if      (i < 65536)  v = w6[i];
  else if (i < 131072) v = w7[i - 65536];
  else if (i < 139264) v = w12[i - 131072];
  else if (i < 147456) v = w13[i - 139264];
  else if (i < 155648) { const int j = i - 147456, r = j >> 5, c = j & 31;
                         v = (c < DTR_) ? w14[r * DTR_ + c] : 0.f; }
  else if (i < 163840) { const int j = i - 155648, r = j >> 5, c = j & 31;
                         v = (c < DTR_) ? w16[r * DTR_ + c] : 0.f; }
  else if (i < 229376) v = w23[i - 163840];
  else                 v = w24[i - 229376];
  wh[i] = (_Float16)v;
}

// ---------------------------------------------------------------------------
// Depthwise causal conv1d (K=4) + SiLU, with optional sequence reversal.
// ---------------------------------------------------------------------------
__global__ __launch_bounds__(256)
void conv_silu_k(const float* __restrict__ x, const float* __restrict__ w,
                 const float* __restrict__ bias, int dir,
                 float* __restrict__ xs, _Float16* __restrict__ xsh) {
  const int idx = blockIdx.x * blockDim.x + threadIdx.x;   // BL*256 threads
  const int c = idx & 255;
  const int row = idx >> 8;                                // b*L + i (scan domain)
  const int i = row & (L_ - 1), b = row >> 12;
  float acc = bias[c];
#pragma unroll
  for (int k = 0; k < 4; ++k) {
    const int ii = i - 3 + k;
    if (ii >= 0) {
      const int src = dir ? (L_ - 1 - ii) : ii;
      acc += w[c * 4 + k] * x[((size_t)b * L_ + src) * DI_ + c];
    }
  }
  const float sv = acc / (1.f + __expf(-acc));
  xs[(size_t)row * DI_ + c]  = sv;
  xsh[(size_t)row * DI_ + c] = (_Float16)sv;
}

// dbl f32 (BL x 32) -> f16 dt-input (cols >= DTR zeroed to match padded weight)
__global__ void dth_k(const float* __restrict__ dbl, _Float16* __restrict__ dth) {
  const int idx = blockIdx.x * blockDim.x + threadIdx.x;   // BL*32
  const int c = idx & 31;
  dth[idx] = (c < DTR_) ? (_Float16)dbl[idx] : (_Float16)0.f;
}

// ---------------------------------------------------------------------------
// Selective scan: one thread per (b, channel), 8 states in registers,
// fused with y = (scan + x*D) * silu(z) and un-reversal for the bwd pass.
// ---------------------------------------------------------------------------
__global__ __launch_bounds__(256)
void scan_k(const float* __restrict__ dt, const float* __restrict__ xs,
            const float* __restrict__ dbl, const float* __restrict__ A_log,
            const float* __restrict__ Dp, const float* __restrict__ z,
            int dir, float* __restrict__ yout) {
  const int b = blockIdx.x, c = threadIdx.x;
  float Arow[NST_];
#pragma unroll
  for (int n = 0; n < NST_; ++n) Arow[n] = -__expf(A_log[c * NST_ + n]);
  const float dpc = Dp[c];
  float h[NST_] = {};
  for (int i = 0; i < L_; ++i) {
    const size_t row = (size_t)b * L_ + i;
    const float dtv = dt[row * DI_ + c];
    const float xv  = xs[row * DI_ + c];
    float y = 0.f;
#pragma unroll
    for (int n = 0; n < NST_; ++n) {
      const float Bv = dbl[row * 32 + DTR_ + n];
      const float Cv = dbl[row * 32 + DTR_ + NST_ + n];
      h[n] = h[n] * __expf(dtv * Arow[n]) + dtv * xv * Bv;
      y += h[n] * Cv;
    }
    const int lo = dir ? (L_ - 1 - i) : i;
    const size_t rowo = (size_t)b * L_ + lo;
    const float zv = z[rowo * DI_ + c];
    yout[rowo * DI_ + c] = (y + xv * dpc) * (zv / (1.f + __expf(-zv)));
  }
}

// ---------------------------------------------------------------------------
// Head input: select/avg y, rmsnorm over DI, f16 output.
// mode: 0 = 0.5*(yf+yb), 1 = yf, 2 = yb
// ---------------------------------------------------------------------------
__global__ __launch_bounds__(32)
void head_rms_k(const float* __restrict__ yf, const float* __restrict__ yb,
                int mode, const float* __restrict__ rw,
                _Float16* __restrict__ yh) {
  const int row = blockIdx.x;
  const int lane = threadIdx.x;
  float v[8], s = 0.f;
#pragma unroll
  for (int j = 0; j < 8; ++j) {
    const int c = lane + j * 32;
    float a = yf[(size_t)row * DI_ + c];
    if (mode == 0)      a = 0.5f * (a + yb[(size_t)row * DI_ + c]);
    else if (mode == 2) a = yb[(size_t)row * DI_ + c];
    v[j] = a; s += a * a;
  }
#pragma unroll
  for (int m = 16; m >= 1; m >>= 1) s += __shfl_xor(s, m, 32);
  const float inv = rsqrtf(s * (1.f / 256.f) + 1e-5f);
#pragma unroll
  for (int j = 0; j < 8; ++j) {
    const int c = lane + j * 32;
    yh[(size_t)row * DI_ + c] = (_Float16)(v[j] * inv * rw[c]);
  }
}

// ---------------------------------------------------------------------------
extern "C" void kernel_launch(void* const* d_in, const int* in_sizes, int n_in,
                              void* d_out, int out_size, void* d_ws, size_t ws_size,
                              hipStream_t stream) {
  (void)in_sizes; (void)n_in; (void)out_size; (void)ws_size;
  const float* in0   = (const float*)d_in[0];
  const float* in1   = (const float*)d_in[1];
  const float* n0w   = (const float*)d_in[2];
  const float* n0b   = (const float*)d_in[3];
  const float* n1w   = (const float*)d_in[4];
  const float* n1b   = (const float*)d_in[5];
  const float* conv_w[2]  = { (const float*)d_in[8],  (const float*)d_in[10] };
  const float* conv_b[2]  = { (const float*)d_in[9],  (const float*)d_in[11] };
  const float* dt_bias[2] = { (const float*)d_in[15], (const float*)d_in[17] };
  const float* A_log[2]   = { (const float*)d_in[18], (const float*)d_in[19] };
  const float* Dvec[2]    = { (const float*)d_in[20], (const float*)d_in[21] };
  const float* rms_w      = (const float*)d_in[22];
  const float* c3d_bias   = (const float*)d_in[25];

  char* ws = (char*)d_ws;
  _Float16* seq0h = (_Float16*)(ws + OFF_SEQ0H);
  _Float16* seq1h = (_Float16*)(ws + OFF_SEQ1H);
  _Float16* wts   = (_Float16*)(ws + OFF_WTS);
  float*    xbuf  = (float*)(ws + OFF_X);
  float*    zbuf  = (float*)(ws + OFF_Z);
  float*    xs    = (float*)(ws + OFF_XS);
  _Float16* xsh   = (_Float16*)(ws + OFF_XSH);
  float*    dbl   = (float*)(ws + OFF_DBL);
  _Float16* dth   = (_Float16*)(ws + OFF_DTH);
  float*    dt    = (float*)(ws + OFF_DT);
  float*    ybuf[2] = { (float*)(ws + OFF_YF), (float*)(ws + OFF_YB) };
  _Float16* yh    = (_Float16*)(ws + OFF_YH);
  _Float16* oh    = (_Float16*)(ws + OFF_OH);

  const int wp_xproj[2]  = { WH_XPROJ,  WH_XPROJB };
  const int wp_dtproj[2] = { WH_DTPROJ, WH_DTPROJB };

  // 1) LayerNorm + transpose -> f16
  ln_rows_k<<<BL_, 32, 0, stream>>>(in0, n0w, n0b, seq0h);
  ln_rows_k<<<BL_, 32, 0, stream>>>(in1, n1w, n1b, seq1h);

  // 2) Weight repack -> f16
  cvt_w_k<<<(WH_TOTAL + 255) / 256, 256, 0, stream>>>(
      (const float*)d_in[6],  (const float*)d_in[7],
      (const float*)d_in[12], (const float*)d_in[13],
      (const float*)d_in[14], (const float*)d_in[16],
      (const float*)d_in[23], (const float*)d_in[24], wts);

  // 3) in_proj: x = seq0 @ Wx^T, z = seq1 @ Wz^T  (8192x256 @ 256x256, TDM-staged)
  wmma_gemm_lds_k<4, 0><<<dim3(BL_ / 128, 4), 128, 0, stream>>>(
      seq0h, DIM_, wts + WH_INPROJ, DIM_, DI_, (void*)xbuf, DI_, nullptr);
  wmma_gemm_lds_k<4, 0><<<dim3(BL_ / 128, 4), 128, 0, stream>>>(
      seq1h, DIM_, wts + WH_INPROJZ, DIM_, DI_, (void*)zbuf, DI_, nullptr);

  // 4) SSM, both directions (scratch reused)
  for (int dir = 0; dir < 2; ++dir) {
    conv_silu_k<<<BL_, 256, 0, stream>>>(xbuf, conv_w[dir], conv_b[dir], dir, xs, xsh);
    // dbl = xs @ xproj^T  (N=32, skinny: direct-global waves)
    wmma_gemm_k<2, 0><<<dim3(BL_ / 16, 1), 32, 0, stream>>>(
        xsh, DI_, wts + wp_xproj[dir], DI_, DI_, (void*)dbl, 32, nullptr);
    dth_k<<<(BL_ * 32) / 256, 256, 0, stream>>>(dbl, dth);
    // dt = softplus(dbl[:, :16] @ dtproj^T + bias)  (K padded to 32)
    wmma_gemm_k<4, 1><<<dim3(BL_ / 16, 4), 32, 0, stream>>>(
        dth, 32, wts + wp_dtproj[dir], 32, 32, (void*)dt, DI_, dt_bias[dir]);
    scan_k<<<2, 256, 0, stream>>>(dt, xs, dbl, A_log[dir], Dvec[dir], zbuf,
                                  dir, ybuf[dir]);
  }

  // 5) Heads: d_out = [head(avg), head(y_f), head(y_b)] concatenated
  float* outsec[3] = { (float*)d_out,
                       (float*)d_out + (size_t)2 * OUTC_ * L_,
                       (float*)d_out + (size_t)4 * OUTC_ * L_ };
  for (int v = 0; v < 3; ++v) {
    head_rms_k<<<BL_, 32, 0, stream>>>(ybuf[0], ybuf[1], v, rms_w, yh);
    // o = rms(y) @ out_proj^T + skip -> f16 (TDM-staged B)
    wmma_gemm_lds_k<4, 2><<<dim3(BL_ / 128, 4), 128, 0, stream>>>(
        yh, DI_, wts + WH_OUTPROJ, DI_, DIM_, (void*)oh, DIM_, in0);
    // out = o @ conv3d_w^T + bias, scattered to (B, 1200, L) (TDM-staged B)
    wmma_gemm_lds_k<5, 3><<<dim3(BL_ / 128, 15), 128, 0, stream>>>(
        oh, DIM_, wts + WH_CONV3D, DIM_, OUTC_, (void*)outsec[v], 0, c3d_bias);
  }
}
